// DBSE_4544075399240
// MI455X (gfx1250) — matrix operand order repaired and checked by reading, not compiled, not run on hardware
//
#include <hip/hip_runtime.h>

// ---------------------------------------------------------------------------
// 2-layer LSTM + reparameterized head on MI455X (gfx1250, wave32).
// f16 WMMA (v_wmma_f32_16x16x32_f16) with fp32 accumulation. Weights are
// pre-packed per launch into WMMA B-fragment order. Per K-tile, the 16
// B-fragments a block needs are staged into LDS with async global->LDS
// copies (ASYNCcnt) and consumed via ds_load_b128, deduplicating the 4x
// B-fragment reuse across waves. Gate nonlinearities fused into the GEMM
// epilogue; h-state ping-pongs per step.
// ---------------------------------------------------------------------------

typedef _Float16 h16;
typedef __attribute__((ext_vector_type(16))) _Float16 v16h;
typedef __attribute__((ext_vector_type(8)))  float    v8f;
typedef __attribute__((ext_vector_type(4)))  int      v4i;

#define WMMA_F16(a, b, c) \
  __builtin_amdgcn_wmma_f32_16x16x32_f16(false, (a), false, (b), (short)0, (c), false, false)

#if defined(__AMDGCN__) && \
    __has_builtin(__builtin_amdgcn_global_load_async_to_lds_b128) && \
    __has_builtin(__builtin_amdgcn_s_wait_asynccnt)
#define USE_ASYNC_LDS 1
#else
#define USE_ASYNC_LDS 0
#endif

constexpr int Bsz = 2048;   // batch
constexpr int Tn  = 64;     // frames
constexpr int Zn  = 64;     // z dim
constexpr int Hn  = 512;    // rnn size

// ---- fragment loaders -----------------------------------------------------
// A (16x32 f16, M x K): lane l -> m = l&15, kbase = (l>=16)?16:0, 16 contiguous k
__device__ __forceinline__ v16h load_a_frag(const h16* __restrict__ A, int ld,
                                            int m, int kcol, int lane) {
  const int kb = (lane >> 4) << 4;
  const h16* r = A + (size_t)m * ld + kcol + kb;
  v16h a;
  ((uint4*)&a)[0] = *(const uint4*)(r);
  ((uint4*)&a)[1] = *(const uint4*)(r + 8);
  return a;
}

// B (32x16 f16) fragment = 512 halves; lane reads halves [lane*16, lane*16+16).
__device__ __forceinline__ v16h load_b_frag(const h16* frag, int lane) {
  const uint4* p = (const uint4*)frag + (lane << 1);
  v16h b;
  ((uint4*)&b)[0] = p[0];
  ((uint4*)&b)[1] = p[1];
  return b;
}

__device__ __forceinline__ float sigm(float x) { return 1.0f / (1.0f + __expf(-x)); }

#if USE_ASYNC_LDS
typedef __attribute__((address_space(1))) v4i* gv4i_p;   // global int4*
typedef __attribute__((address_space(3))) v4i* lv4i_p;   // LDS int4*
typedef __attribute__((address_space(1))) void* gvoid_p;
typedef __attribute__((address_space(3))) void* lvoid_p;

// Stage the 16 B-fragments (16KB) of k-tile `kt` into LDS buffer `dst`.
// 256 threads x 64B each; 4 async b128 ops per thread -> ASYNCcnt += 4/wave.
template <int KT>
__device__ __forceinline__ void stage_b(const h16* __restrict__ Wp, int ntb4,
                                        int kt, h16* dst, int tid) {
  const int fl = tid >> 4;                                  // local frag 0..15
  const int nt = ((fl >> 2) << 5) + ntb4 + (fl & 3);        // gate*32 + by*4 + j
  const h16* src = Wp + (((size_t)nt * KT + kt) << 9) + ((tid & 15) << 5);
  h16* d = dst + (fl << 9) + ((tid & 15) << 5);
#pragma unroll
  for (int q = 0; q < 4; ++q) {
    __builtin_amdgcn_global_load_async_to_lds_b128(
        (gv4i_p)(gvoid_p)(src + q * 8),
        (lv4i_p)(lvoid_p)(d + q * 8), 0, 0);
  }
}
#endif

// ---- fused GEMM + LSTM cell ----------------------------------------------
// gates[B,2048] = [A0|A1] @ Wp + bias, then c/h update.
// Block: 64 rows x 64 hidden units (=> 64 x 256 gate cols). 8 waves:
//   wave w -> rows 16*(w&3), hidden cols 32*(w>>2), all 4 gates.
template <int NKT0, int NKT1, int LDA0>
__global__ __launch_bounds__(256)
void lstm_cell_gemm(const h16* __restrict__ A0, const h16* __restrict__ A1,
                    const h16* __restrict__ Wp,
                    const float* __restrict__ bias,
                    float* __restrict__ c_state,
                    h16* __restrict__ h_out) {
  constexpr int KT = NKT0 + NKT1;
  const int tid  = threadIdx.x;
  const int lane = tid & 31;
  const int wid  = tid >> 5;
  const int mrow = blockIdx.x * 64 + ((wid & 3) << 4);
  const int hidb = blockIdx.y * 64 + ((wid >> 2) << 5);
  const int ntb  = hidb >> 4;
  const int m_l  = mrow + (lane & 15);

  const v8f vzero = {};
  v8f acc[4][2];
#pragma unroll
  for (int g = 0; g < 4; ++g) { acc[g][0] = vzero; acc[g][1] = vzero; }

#if USE_ASYNC_LDS
  __shared__ h16 bsm[2][16 * 512];          // 2 x 16KB double buffer
  const int ntb4 = blockIdx.y << 2;
  const int wgrp = (wid >> 2) << 1;         // wave's j-base within frag group
  stage_b<KT>(Wp, ntb4, 0, bsm[0], tid);
  stage_b<KT>(Wp, ntb4, 1, bsm[1], tid);
#endif

#pragma unroll 1
  for (int seg = 0; seg < 2; ++seg) {
    const h16* A  = seg ? A1 : A0;
    const int  ld = seg ? Hn : LDA0;
    const int  nk = seg ? NKT1 : NKT0;
    const int  ko = seg ? NKT0 : 0;
#pragma unroll 1
    for (int kt = 0; kt < nk; ++kt) {
      const int  ktg = ko + kt;
      const v16h a   = load_a_frag(A, ld, m_l, kt << 5, lane);

#if USE_ASYNC_LDS
      if (ktg + 1 < KT) __builtin_amdgcn_s_wait_asynccnt(4);   // stage ktg done
      else              __builtin_amdgcn_s_wait_asynccnt(0);   // last stage
      __syncthreads();
      const h16* bbuf = bsm[ktg & 1];
#pragma unroll
      for (int g = 0; g < 4; ++g) {
#pragma unroll
        for (int s = 0; s < 2; ++s) {
          const int fl = (g << 2) + wgrp + s;
          const v16h b = load_b_frag(bbuf + (fl << 9), lane);  // ds_load_b128 x2
          acc[g][s] = WMMA_F16(a, b, acc[g][s]);
        }
      }
      __syncthreads();                                         // all readers done
      if (ktg + 2 < KT) stage_b<KT>(Wp, ntb4, ktg + 2, bsm[ktg & 1], tid);
#else
#pragma unroll
      for (int g = 0; g < 4; ++g) {
#pragma unroll
        for (int s = 0; s < 2; ++s) {
          const int nt = (g << 5) + ntb + s;   // nt*KT folds to immediates (KT const)
          const v16h b = load_b_frag(Wp + (((size_t)nt * KT + ktg) << 9), lane);
          acc[g][s] = WMMA_F16(a, b, acc[g][s]);
        }
      }
#endif
    }
  }

  // Epilogue: bias + LSTM cell; all four gates live in this wave's registers.
#pragma unroll
  for (int s = 0; s < 2; ++s) {
    const int col = hidb + (s << 4) + (lane & 15);
    const float bi = bias[col];
    const float bf = bias[Hn + col];
    const float bg = bias[2 * Hn + col];
    const float bo = bias[3 * Hn + col];
#pragma unroll
    for (int v = 0; v < 8; ++v) {
      const int row = mrow + v + ((lane >> 4) << 3);
      const size_t idx = (size_t)row * Hn + col;
      const float ig = sigm(acc[0][s][v] + bi);
      const float fg = sigm(acc[1][s][v] + bf);
      const float gg = tanhf(acc[2][s][v] + bg);
      const float og = sigm(acc[3][s][v] + bo);
      const float cn = fg * c_state[idx] + ig * gg;
      c_state[idx] = cn;
      h_out[idx]   = (h16)(og * tanhf(cn));
    }
  }
}

// ---- output head: [B,128] = h2 @ [Wmean|Wlogvar]^T, then reparameterize ---
__global__ __launch_bounds__(256)
void out_proj(const h16* __restrict__ h2, const h16* __restrict__ Wo,
              const float* __restrict__ bo, const float* __restrict__ eps, int t,
              float* __restrict__ zm, float* __restrict__ zl, float* __restrict__ zo) {
  __shared__ float smem[64][130];                 // [row][mean 0..63 | logvar 64..127]
  const int lane = threadIdx.x & 31;
  const int wid  = threadIdx.x >> 5;
  const int mrow = blockIdx.x * 64 + ((wid & 3) << 4);
  const int ncb  = (wid >> 2) << 6;               // 0 (mean waves) or 64 (logvar waves)

  const v8f vzero = {};
  v8f acc[4] = {vzero, vzero, vzero, vzero};
  const int m_l = mrow + (lane & 15);

  for (int kt = 0; kt < 16; ++kt) {               // K = 512
    const v16h a = load_a_frag(h2, Hn, m_l, kt << 5, lane);
#pragma unroll
    for (int s = 0; s < 4; ++s) {
      const int nt = (ncb >> 4) + s;
      const v16h b = load_b_frag(Wo + (((size_t)nt * 16 + kt) << 9), lane);
      acc[s] = WMMA_F16(a, b, acc[s]);
    }
  }

#pragma unroll
  for (int s = 0; s < 4; ++s) {
    const int n = ncb + (s << 4) + (lane & 15);
    const float bv = bo[n];
#pragma unroll
    for (int v = 0; v < 8; ++v) {
      const int rl = ((wid & 3) << 4) + v + ((lane >> 4) << 3);
      smem[rl][n] = acc[s][v] + bv;
    }
  }
  __syncthreads();

  const int rowbase = blockIdx.x * 64;
  for (int e = threadIdx.x; e < 64 * 64; e += 256) {
    const int rl = e >> 6, zc = e & 63;
    const float m  = smem[rl][zc];
    const float lv = smem[rl][64 + zc];
    const size_t gi = ((size_t)(rowbase + rl) * Tn + t) * Zn + zc;
    zm[gi] = m;
    zl[gi] = lv;
    zo[gi] = m + eps[gi] * __expf(0.5f * lv);
  }
}

// ---- once-per-launch prep kernels ----------------------------------------
// Pack W^T = [Wa^T ; Wb^T] (K = Ka+Kb, N = NT*16) into B-fragment order:
// out[(nt*KT+kt)*512 + lane*16 + i] = W^T[kt*32 + (lane>=16)*16 + i][nt*16 + (lane&15)]
__global__ void pack_w(const float* __restrict__ Wa, int Ka,
                       const float* __restrict__ Wb, int Kb,
                       h16* __restrict__ out, int KT, int NT) {
  const int p = blockIdx.x * blockDim.x + threadIdx.x;
  const int total = NT * KT * 512;
  if (p >= total) return;
  const int frag = p >> 9, r = p & 511;
  const int lane = r >> 4, i = r & 15;
  const int nt = frag / KT, kt = frag % KT;
  const int n = (nt << 4) + (lane & 15);
  const int k = (kt << 5) + ((lane >> 4) << 4) + i;
  const float v = (k < Ka) ? Wa[(size_t)n * Ka + k] : Wb[(size_t)n * Kb + (k - Ka)];
  out[p] = (h16)v;
}

__global__ void pack_wout(const float* __restrict__ Wm, const float* __restrict__ Wl,
                          h16* __restrict__ out) {
  const int p = blockIdx.x * blockDim.x + threadIdx.x;
  if (p >= 8 * 16 * 512) return;
  const int frag = p >> 9, r = p & 511;
  const int lane = r >> 4, i = r & 15;
  const int nt = frag / 16, kt = frag % 16;
  const int n = (nt << 4) + (lane & 15);
  const int k = (kt << 5) + ((lane >> 4) << 4) + i;
  const float v = (n < 64) ? Wm[(size_t)n * Hn + k] : Wl[(size_t)(n - 64) * Hn + k];
  out[p] = (h16)v;
}

// x_t = z_post[:, t-1] (x_0 = 0), converted to f16, time-major [T][B][Z].
__global__ void build_x(const float* __restrict__ zp, h16* __restrict__ x) {
  const int idx = blockIdx.x * blockDim.x + threadIdx.x;
  if (idx >= Tn * Bsz * Zn) return;
  const int t = idx / (Bsz * Zn);
  const int rem = idx % (Bsz * Zn);
  const int b = rem >> 6, z = rem & 63;
  const float v = (t == 0) ? 0.0f : zp[((size_t)b * Tn + (t - 1)) * Zn + z];
  x[idx] = (h16)v;
}

__global__ void build_bias(const float* bi1, const float* bh1,
                           const float* bi2, const float* bh2,
                           const float* bm, const float* bl,
                           float* b1, float* b2, float* bo) {
  const int i = blockIdx.x * blockDim.x + threadIdx.x;
  if (i < 4 * Hn) { b1[i] = bi1[i] + bh1[i]; b2[i] = bi2[i] + bh2[i]; }
  if (i < 128)    { bo[i] = (i < 64) ? bm[i] : bl[i - 64]; }
}

__global__ void init_state(h16* h1, h16* h2, float* c1, float* c2) {
  const int i = blockIdx.x * blockDim.x + threadIdx.x;
  if (i < Bsz * Hn) { h1[i] = (h16)0.0f; h2[i] = (h16)0.0f; c1[i] = 0.0f; c2[i] = 0.0f; }
}

// ---------------------------------------------------------------------------
extern "C" void kernel_launch(void* const* d_in, const int* in_sizes, int n_in,
                              void* d_out, int out_size, void* d_ws, size_t ws_size,
                              hipStream_t stream) {
  (void)in_sizes; (void)n_in; (void)out_size; (void)ws_size;

  const float* z_post = (const float*)d_in[0];
  const float* eps    = (const float*)d_in[1];
  const float* W_ih1  = (const float*)d_in[2];
  const float* W_hh1  = (const float*)d_in[3];
  const float* b_ih1  = (const float*)d_in[4];
  const float* b_hh1  = (const float*)d_in[5];
  const float* W_ih2  = (const float*)d_in[6];
  const float* W_hh2  = (const float*)d_in[7];
  const float* b_ih2  = (const float*)d_in[8];
  const float* b_hh2  = (const float*)d_in[9];
  const float* W_mean = (const float*)d_in[10];
  const float* b_mean = (const float*)d_in[11];
  const float* W_logv = (const float*)d_in[12];
  const float* b_logv = (const float*)d_in[13];

  float* zm = (float*)d_out;
  float* zl = zm + (size_t)Bsz * Tn * Zn;
  float* zo = zl + (size_t)Bsz * Tn * Zn;

  // workspace carve-out (256B aligned)
  char* wp = (char*)d_ws;
  auto take = [&](size_t bytes) -> char* {
    char* r = wp; wp += (bytes + 255) & ~(size_t)255; return r;
  };
  h16*   xF   = (h16*)  take((size_t)Tn * Bsz * Zn * 2);   // 16 MB
  h16*   pW1  = (h16*)  take((size_t)128 * 18 * 512 * 2);  // K=576,  N=2048
  h16*   pW2  = (h16*)  take((size_t)128 * 32 * 512 * 2);  // K=1024, N=2048
  h16*   pWo  = (h16*)  take((size_t)8 * 16 * 512 * 2);    // K=512,  N=128
  float* b1   = (float*)take(4 * Hn * 4);
  float* b2   = (float*)take(4 * Hn * 4);
  float* bo   = (float*)take(128 * 4);
  h16*   h1b0 = (h16*)  take((size_t)Bsz * Hn * 2);
  h16*   h1b1 = (h16*)  take((size_t)Bsz * Hn * 2);
  h16*   h2b0 = (h16*)  take((size_t)Bsz * Hn * 2);
  h16*   h2b1 = (h16*)  take((size_t)Bsz * Hn * 2);
  float* c1   = (float*)take((size_t)Bsz * Hn * 4);
  float* c2   = (float*)take((size_t)Bsz * Hn * 4);
  h16* h1b[2] = {h1b0, h1b1};
  h16* h2b[2] = {h2b0, h2b1};

  // once-per-launch prep (deterministic, runs every call)
  build_x<<<(Tn * Bsz * Zn + 255) / 256, 256, 0, stream>>>(z_post, xF);
  pack_w<<<(128 * 18 * 512 + 255) / 256, 256, 0, stream>>>(W_ih1, Zn, W_hh1, Hn, pW1, 18, 128);
  pack_w<<<(128 * 32 * 512 + 255) / 256, 256, 0, stream>>>(W_ih2, Hn, W_hh2, Hn, pW2, 32, 128);
  pack_wout<<<(8 * 16 * 512 + 255) / 256, 256, 0, stream>>>(W_mean, W_logv, pWo);
  build_bias<<<(4 * Hn + 255) / 256, 256, 0, stream>>>(b_ih1, b_hh1, b_ih2, b_hh2,
                                                       b_mean, b_logv, b1, b2, bo);
  init_state<<<(Bsz * Hn + 255) / 256, 256, 0, stream>>>(h1b[0], h2b[0], c1, c2);

  const dim3 cgrid(Bsz / 64, Hn / 64);  // 32 x 8 blocks, 256 thr (8 waves)
  for (int t = 0; t < Tn; ++t) {
    const int cur = t & 1, nxt = (t + 1) & 1;
    // layer 1: A = [x_t (K=64) | h1 (K=512)]
    lstm_cell_gemm<2, 16, Zn><<<cgrid, 256, 0, stream>>>(
        xF + (size_t)t * Bsz * Zn, h1b[cur], pW1, b1, c1, h1b[nxt]);
    // layer 2: A = [h1_new (K=512) | h2 (K=512)]
    lstm_cell_gemm<16, 16, Hn><<<cgrid, 256, 0, stream>>>(
        h1b[nxt], h2b[cur], pW2, b2, c2, h2b[nxt]);
    // head: mean/logvar + reparameterized sample
    out_proj<<<Bsz / 64, 256, 0, stream>>>(h2b[nxt], pWo, bo, eps, t, zm, zl, zo);
  }
}
